// DecisionStep_38878043964009
// MI455X (gfx1250) — matrix operand order repaired and checked by reading, not compiled, not run on hardware
//
#include <hip/hip_runtime.h>
#include <hip/hip_bf16.h>

// TabNet decision step on gfx1250 (MI455X).
//   - GEMMs: bf16 WMMA (v_wmma_f32_16x16x32_bf16), f32 accumulate (TF32-style).
//   - GhostBN fused into GEMM epilogue (VBS=128 == M-tile, full stats in-block).
//   - GLU gate fused (each block owns col strip j and j+1024).
//   - All f32->bf16 conversion hoisted OUT of the GEMM: weights pre-transposed
//     to bf16 [N][K] once per launch; activations emitted bf16 by the producer.
//   - Tile staging: GLOBAL_LOAD_ASYNC_TO_LDS_B128 (ASYNCcnt), double-buffered:
//     tile k+1 prefetch stays in flight behind tile k's WMMAs; in-order
//     ASYNCcnt retirement lets "s_wait_asynccnt <per-tile-count>" release
//     exactly the tile about to be consumed.

typedef __attribute__((ext_vector_type(16))) __bf16 v16bf;
typedef __attribute__((ext_vector_type(8)))  __bf16 bf16x8;
typedef __attribute__((ext_vector_type(4)))  __bf16 bf16x4;
typedef __attribute__((ext_vector_type(8)))  float  v8f;

#define NB      16384
#define NINP    1024
#define NNA     512
#define NOD     1024

__device__ __forceinline__ unsigned lds_off(const void* p)
{
    // LDS aperture: generic addr[31:0] == LDS byte offset (ISA 10.2)
    return (unsigned)(uintptr_t)p;
}

__device__ __forceinline__ void async_copy16(unsigned lds_byte, const void* gptr)
{
    asm volatile("global_load_async_to_lds_b128 %0, %1, off"
                 :: "v"(lds_byte), "v"((unsigned long long)(uintptr_t)gptr)
                 : "memory");
}

template<int N>
__device__ __forceinline__ void wait_async()
{
    asm volatile("s_wait_asynccnt %0" :: "n"(N) : "memory");
}

// ---------------------------------------------------------------------------
// Fused GEMM + GhostBN (+ GLU + residual).
//   z = A[B x K](bf16) @ Wt[N x K](bf16)^T + bias -> GBN(128-row ghost batch)
//   DO_GLU: out = z1*sigmoid(z2) (+resid)*sqrt(0.5); writes f32 + bf16 copies.
// Block = 256 threads (8 waves); tile 128 rows x (HALVES x 128) cols.
// ---------------------------------------------------------------------------
template<int HALVES, bool DO_GLU, bool RESID>
__global__ __launch_bounds__(256)
void gemm_gbn_glu(const __bf16* __restrict__ A,    // [B x K] row-major bf16
                  const __bf16* __restrict__ Wt,   // [N x K] bf16 (pre-transposed)
                  const float* __restrict__ bias,
                  const float* __restrict__ gamma,
                  const float* __restrict__ beta,
                  const float* __restrict__ resid, // [B x 1024] f32 or null
                  float* __restrict__ out,
                  __bf16* __restrict__ out_bf,     // bf16 copy for next layer A
                  int K, int N)
{
    constexpr int APT = 2 + 2 * HALVES;                  // async ops per tile/wave
    __shared__ alignas(32) __bf16 lA[2][128][32];        // [buf][row][k]
    __shared__ alignas(32) __bf16 lB[2][HALVES][128][32];// [buf][col][k]
    __shared__ float rSum[HALVES][128];
    __shared__ float rSsq[HALVES][128];

    const int tid   = threadIdx.x;
    const int wave  = tid >> 5;
    const int lane  = tid & 31;
    const int l16   = lane & 15;
    const int kh    = lane >> 4;
    const int strip = blockIdx.x;          // 0..7
    const int rowbase = blockIdx.y * 128;  // == ghost batch

    const unsigned lAo = lds_off(&lA[0][0][0]);
    const unsigned lBo = lds_off(&lB[0][0][0][0]);

    // ---- issue one K-tile's async global->LDS copies (16B granules) --------
    auto stage = [&](int k0, int buf) {
#pragma unroll
        for (int i = 0; i < 2; ++i) {
            const int cidx = tid + 256 * i;
            const int r = cidx >> 2, c8 = (cidx & 3) << 3;
            async_copy16(lAo + buf * 8192 + r * 64 + c8 * 2,
                         A + (size_t)(rowbase + r) * K + k0 + c8);
        }
#pragma unroll
        for (int h = 0; h < HALVES; ++h) {
            const int colbase = strip * 128 + h * 1024;
#pragma unroll
            for (int i = 0; i < 2; ++i) {
                const int cidx = tid + 256 * i;
                const int c = cidx >> 2, k8 = (cidx & 3) << 3;
                async_copy16(lBo + (buf * HALVES + h) * 8192 + c * 64 + k8 * 2,
                             Wt + (size_t)(colbase + c) * K + k0 + k8);
            }
        }
    };

    const v8f vzero = {0.f,0.f,0.f,0.f,0.f,0.f,0.f,0.f};
    v8f acc[HALVES][8];
#pragma unroll
    for (int h = 0; h < HALVES; ++h)
#pragma unroll
        for (int t = 0; t < 8; ++t) acc[h][t] = vzero;

    const int iters = K >> 5;
    stage(0, 0);
    for (int k = 0; k < iters; ++k) {
        const int buf = k & 1;
        if (k + 1 < iters) {
            stage((k + 1) << 5, buf ^ 1);  // prefetch stays behind this tile
            wait_async<APT>();             // in-order: tile k's APT copies done
        } else {
            wait_async<0>();
        }
        __syncthreads();

        // ---- A fragment (contiguous k order, same as B): lanes0-15 K{0-15},
        //      lanes16-31 K{16-31}, row M = 16*wave + l16 ----
        const __bf16* pa = &lA[buf][wave * 16 + l16][kh * 16];
        bf16x8 alo = *(const bf16x8*)pa;
        bf16x8 ahi = *(const bf16x8*)(pa + 8);
        v16bf afrag = __builtin_shufflevector(alo, ahi,
            0,1,2,3,4,5,6,7,8,9,10,11,12,13,14,15);

#pragma unroll
        for (int h = 0; h < HALVES; ++h) {
#pragma unroll
            for (int t = 0; t < 8; ++t) {
                const v16bf* pb = (const v16bf*)&lB[buf][h][t * 16 + l16][kh * 16];
                acc[h][t] = __builtin_amdgcn_wmma_f32_16x16x32_bf16(
                    false, afrag, false, *pb, (short)0, acc[h][t], false, false);
            }
        }
        __syncthreads();   // all waves done reading buf before its reuse (k+2)
    }

    // ---------------- GhostBN epilogue -------------------------------------
    {
        float* rs = &rSum[0][0];
        float* rq = &rSsq[0][0];
        for (int i = tid; i < HALVES * 128; i += 256) { rs[i] = 0.f; rq[i] = 0.f; }
    }
    __syncthreads();
#pragma unroll
    for (int h = 0; h < HALVES; ++h) {
#pragma unroll
        for (int t = 0; t < 8; ++t) {
            const int lc = t * 16 + l16;
            const int gcol = strip * 128 + lc + h * 1024;
            const float bv = bias[gcol];
            float s = 0.f, q = 0.f;
#pragma unroll
            for (int r = 0; r < 8; ++r) {
                float v = acc[h][t][r] + bv;
                acc[h][t][r] = v;
                s += v; q += v * v;
            }
            atomicAdd(&rSum[h][lc], s);   // ds_add_f32
            atomicAdd(&rSsq[h][lc], q);
        }
    }
    __syncthreads();
#pragma unroll
    for (int h = 0; h < HALVES; ++h) {
#pragma unroll
        for (int t = 0; t < 8; ++t) {
            const int lc = t * 16 + l16;
            const int gcol = strip * 128 + lc + h * 1024;
            const float mean = rSum[h][lc] * (1.f / 128.f);
            const float var  = rSsq[h][lc] * (1.f / 128.f) - mean * mean;
            const float rstd = rsqrtf(var + 1e-5f);
            const float g = gamma[gcol], bb = beta[gcol];
#pragma unroll
            for (int r = 0; r < 8; ++r)
                acc[h][t][r] = (acc[h][t][r] - mean) * rstd * g + bb;
        }
    }

    // ---------------- store (C layout: lane=col, vgpr=row, +8 hi lanes) -----
    const int m0 = rowbase + wave * 16 + kh * 8;
#pragma unroll
    for (int t = 0; t < 8; ++t) {
        const int lc  = t * 16 + l16;
        const int col = strip * 128 + lc;
#pragma unroll
        for (int r = 0; r < 8; ++r) {
            const int row = m0 + r;
            if constexpr (DO_GLU) {
                float z1 = acc[0][t][r];
                float z2 = acc[1][t][r];
                float v = z1 * (1.f / (1.f + __expf(-z2)));
                if constexpr (RESID)
                    v = (resid[(size_t)row * 1024 + col] + v) * 0.70710678118654752f;
                out[(size_t)row * 1024 + col]    = v;
                out_bf[(size_t)row * 1024 + col] = (__bf16)v;
            } else {
                out[(size_t)row * N + col] = acc[0][t][r];
            }
        }
    }
}

// ---------------------------------------------------------------------------
// Sparsemax (binary search of tau), per-wave row of 1024 features.
// Emits new_priors (f32), masked input x*mask (bf16, next GEMM A operand),
// and atomically accumulates the sparse-entropy loss.
// ---------------------------------------------------------------------------
__global__ __launch_bounds__(256)
void sparsemax_kernel(const float* __restrict__ a1,
                      const float* __restrict__ x,
                      const float* __restrict__ priors,
                      __bf16* __restrict__ xm,
                      float* __restrict__ new_priors,
                      float* __restrict__ loss)
{
    const int wave = threadIdx.x >> 5;
    const int lane = threadIdx.x & 31;
    const int row  = blockIdx.x * 8 + wave;
    const size_t base = (size_t)row * NINP;

    float z[32];
    float mx = -3.4e38f;
#pragma unroll
    for (int j = 0; j < 32; ++j) {
        z[j] = a1[base + lane + 32 * j] * priors[base + lane + 32 * j];
        mx = fmaxf(mx, z[j]);
    }
#pragma unroll
    for (int off = 16; off; off >>= 1) mx = fmaxf(mx, __shfl_xor(mx, off, 32));
#pragma unroll
    for (int j = 0; j < 32; ++j) z[j] -= mx;

    float lo = -1.0f, hi = 0.0f;
    for (int it = 0; it < 28; ++it) {
        const float tau = 0.5f * (lo + hi);
        float s = 0.f;
#pragma unroll
        for (int j = 0; j < 32; ++j) s += fmaxf(0.f, z[j] - tau);
#pragma unroll
        for (int off = 16; off; off >>= 1) s += __shfl_xor(s, off, 32);
        if (s > 1.0f) lo = tau; else hi = tau;
    }
    const float tau = 0.5f * (lo + hi);

    float lsum = 0.f;
#pragma unroll
    for (int j = 0; j < 32; ++j) {
        const int i = lane + 32 * j;
        const float m = fmaxf(0.f, z[j] - tau);
        new_priors[base + i] = priors[base + i] * (1.5f - m);
        xm[base + i]         = (__bf16)(x[base + i] * m);
        lsum += -m * __logf(m + 1e-10f);
    }
#pragma unroll
    for (int off = 16; off; off >>= 1) lsum += __shfl_xor(lsum, off, 32);
    if (lane == 0)
        atomicAdd(loss, lsum * (1.0f / ((float)NB * (float)NINP)));
}

// ---------------------------------------------------------------------------
// One-shot prep kernels: f32->bf16 convert, and convert+transpose for weights.
// ---------------------------------------------------------------------------
__global__ __launch_bounds__(256)
void cvt_bf16(const float* __restrict__ in, __bf16* __restrict__ out, int n)
{
    const int i = (blockIdx.x * 256 + threadIdx.x) * 4;
    if (i + 3 < n) {
        const float4 v = *(const float4*)(in + i);
        bf16x4 o = { (__bf16)v.x, (__bf16)v.y, (__bf16)v.z, (__bf16)v.w };
        *(bf16x4*)(out + i) = o;
    }
}

__global__ __launch_bounds__(256)
void transpose_bf16(const float* __restrict__ W, __bf16* __restrict__ Wt,
                    int K, int N)   // W[K][N] f32 -> Wt[N][K] bf16
{
    __shared__ float tile[32][33];
    const int k0 = blockIdx.x * 32, n0 = blockIdx.y * 32;
    const int tx = threadIdx.x & 31, ty = threadIdx.x >> 5;  // ty: 0..7
#pragma unroll
    for (int r = ty; r < 32; r += 8)
        tile[r][tx] = W[(size_t)(k0 + r) * N + n0 + tx];
    __syncthreads();
#pragma unroll
    for (int r = ty; r < 32; r += 8)
        Wt[(size_t)(n0 + r) * K + k0 + tx] = (__bf16)tile[tx][r];
}

__global__ void zero_scalar(float* p)
{
    if (threadIdx.x == 0 && blockIdx.x == 0) *p = 0.f;
}

// ---------------------------------------------------------------------------
extern "C" void kernel_launch(void* const* d_in, const int* in_sizes, int n_in,
                              void* d_out, int out_size, void* d_ws, size_t ws_size,
                              hipStream_t stream)
{
    const float* x      = (const float*)d_in[0];
    const float* a      = (const float*)d_in[1];
    const float* priors = (const float*)d_in[2];
    const float* att_w  = (const float*)d_in[3];
    const float* att_b  = (const float*)d_in[4];
    const float* att_g  = (const float*)d_in[5];
    const float* att_bb = (const float*)d_in[6];
    const float* s0_w = (const float*)d_in[7];  const float* s0_b = (const float*)d_in[8];
    const float* s0_g = (const float*)d_in[9];  const float* s0_bb = (const float*)d_in[10];
    const float* s1_w = (const float*)d_in[11]; const float* s1_b = (const float*)d_in[12];
    const float* s1_g = (const float*)d_in[13]; const float* s1_bb = (const float*)d_in[14];
    const float* i0_w = (const float*)d_in[15]; const float* i0_b = (const float*)d_in[16];
    const float* i0_g = (const float*)d_in[17]; const float* i0_bb = (const float*)d_in[18];
    const float* i1_w = (const float*)d_in[19]; const float* i1_b = (const float*)d_in[20];
    const float* i1_g = (const float*)d_in[21]; const float* i1_bb = (const float*)d_in[22];

    float* out_h    = (float*)d_out;
    float* out_loss = out_h + (size_t)NB * NOD;
    float* out_np   = out_loss + 1;

    // ---- workspace layout ----
    const size_t M = (size_t)NB * 1024;           // 16M elements
    float*  F0    = (float*)d_ws;                 // h f32 ping
    float*  F1    = F0 + M;                       // a1, then h f32 pong
    __bf16* Hb0   = (__bf16*)(F1 + M);            // h bf16 ping
    __bf16* Hb1   = Hb0 + M;                      // h bf16 pong
    __bf16* Abf   = Hb1 + M;                      // bf16(a)     [B x 512]
    __bf16* XMb   = Abf + (size_t)NB * NNA;       // bf16(x*mask)[B x 1024]
    __bf16* WTatt = XMb + M;                      // [1024 x 512]
    __bf16* WTs0  = WTatt + (size_t)NINP * NNA;   // [2048 x 1024] each below
    __bf16* WTs1  = WTs0 + (size_t)2048 * 1024;
    __bf16* WTi0  = WTs1 + (size_t)2048 * 1024;
    __bf16* WTi1  = WTi0 + (size_t)2048 * 1024;

    const dim3 blk(256);
    const dim3 ggrid(8, NB / 128);

    // 0) one-shot conversions (weights f32[K][N] -> bf16[N][K]; a -> bf16)
    transpose_bf16<<<dim3(NNA / 32, NINP / 32), blk, 0, stream>>>(att_w, WTatt, NNA, NINP);
    transpose_bf16<<<dim3(NINP / 32, 2048 / 32), blk, 0, stream>>>(s0_w, WTs0, NINP, 2048);
    transpose_bf16<<<dim3(NOD / 32, 2048 / 32), blk, 0, stream>>>(s1_w, WTs1, NOD, 2048);
    transpose_bf16<<<dim3(NOD / 32, 2048 / 32), blk, 0, stream>>>(i0_w, WTi0, NOD, 2048);
    transpose_bf16<<<dim3(NOD / 32, 2048 / 32), blk, 0, stream>>>(i1_w, WTi1, NOD, 2048);
    cvt_bf16<<<(NB * NNA) / 1024, blk, 0, stream>>>(a, Abf, NB * NNA);

    // 1) attention: a1 = GBN(a @ att_w + b)   -> F1 (f32)
    gemm_gbn_glu<1, false, false><<<ggrid, blk, 0, stream>>>(
        Abf, WTatt, att_b, att_g, att_bb, nullptr, F1, nullptr, NNA, NINP);

    // 2) sparsemax + new_priors + loss + bf16 masked input
    zero_scalar<<<1, 64, 0, stream>>>(out_loss);
    sparsemax_kernel<<<NB / 8, blk, 0, stream>>>(F1, x, priors, XMb, out_np, out_loss);

    // 3) feature transformer: 4 fused GLU blocks (ping-pong f32 + bf16)
    gemm_gbn_glu<2, true, false><<<ggrid, blk, 0, stream>>>(
        XMb, WTs0, s0_b, s0_g, s0_bb, nullptr, F0, Hb0, NINP, 2048);
    gemm_gbn_glu<2, true, true><<<ggrid, blk, 0, stream>>>(
        Hb0, WTs1, s1_b, s1_g, s1_bb, F0, F1, Hb1, NOD, 2048);
    gemm_gbn_glu<2, true, true><<<ggrid, blk, 0, stream>>>(
        Hb1, WTi0, i0_b, i0_g, i0_bb, F1, F0, Hb0, NOD, 2048);
    gemm_gbn_glu<2, true, true><<<ggrid, blk, 0, stream>>>(
        Hb0, WTi1, i1_b, i1_g, i1_bb, F0, out_h, Hb1, NOD, 2048);
}
